// ExpMax_62654982914638
// MI455X (gfx1250) — matrix operand order repaired and checked by reading, not compiled, not run on hardware
//
#include <hip/hip_runtime.h>

// ---------------- problem constants ----------------
#define NPTS   262144
#define KCL    512
#define DIM    64
#define TN     64        // rows per tile
#define NSTEPS 5
#define GRID_MAIN 512
#define TPB    512       // 16 wave32s

// ---------------- vector types ----------------
typedef __attribute__((ext_vector_type(16))) __bf16         v16bf;
typedef __attribute__((ext_vector_type(16))) unsigned short v16us;
typedef __attribute__((ext_vector_type(8)))  float          v8f;
typedef __attribute__((ext_vector_type(8)))  unsigned int   v8u;
typedef __attribute__((ext_vector_type(4)))  unsigned int   u32x4;
typedef __attribute__((ext_vector_type(4)))  float          f32x4;

// ---------------- bf16 split helpers (truncation; residual absorbs error) --
__device__ __forceinline__ unsigned short bf16_trunc(float f) {
    return (unsigned short)(__float_as_uint(f) >> 16);
}
__device__ __forceinline__ float bf16_f32(unsigned short h) {
    return __uint_as_float(((unsigned int)h) << 16);
}
// pack {hi in [15:0], lo-residual in [31:16]}
__device__ __forceinline__ unsigned int pack_split(float x) {
    unsigned short h = bf16_trunc(x);
    unsigned short l = bf16_trunc(x - bf16_f32(h));
    return (unsigned int)h | ((unsigned int)l << 16);
}

// ---------------- LDS layout (bytes); 293632 B <= 320 KB WGP budget --------
#define LOGITS_STRIDE 516         // 512 + 4 pad (16B-aligned rows, bank spread)
#define RESPT_STRIDE  68          // 64 + 4 pad, keeps 16B alignment for b128
#define DATAT_STRIDE  68
#define L_LOGITS 0                // 64*516*4  = 132096
#define L_RESPT  132096           // 512*68*4  = 139264
#define L_DATAT  271360           // 64*68*4   = 17408
#define L_C2     288768           // 512*4
#define L_IVN    290816           // 64*4
#define L_MROW   291072           // 64*4
#define L_RROW   291328           // 64*4
#define L_RED    291584           // 64*8*4
#define LDS_BYTES 293632

// ===========================================================================
// Fused EM step: logits GEMM (bf16 hi/lo split, 3 WMMAs) -> softmax in LDS ->
// transposed resp -> update GEMM (3 WMMAs) with register-persistent num tiles.
// ===========================================================================
__global__ __launch_bounds__(TPB, 1)
void em_main(const float* __restrict__ data,
             const float* __restrict__ var_noise,
             const unsigned short* __restrict__ cent_hi,
             const unsigned short* __restrict__ cent_lo,
             const float* __restrict__ c2g,
             float* __restrict__ num,
             float* __restrict__ den)
{
    extern __shared__ char smem[];
    float*        logits = (float*)(smem + L_LOGITS);
    unsigned int* respT  = (unsigned int*)(smem + L_RESPT);
    unsigned int* dataT  = (unsigned int*)(smem + L_DATAT);
    float*        c2s    = (float*)(smem + L_C2);
    float*        ivn    = (float*)(smem + L_IVN);
    float*        mrow   = (float*)(smem + L_MROW);
    float*        rrow   = (float*)(smem + L_RROW);
    float*        red    = (float*)(smem + L_RED);

    const int t    = threadIdx.x;
    const int wave = t >> 5;
    const int lane = t & 31;
    const int m    = lane & 15;    // column / row-in-tile index
    const int lh   = lane >> 4;    // lane half (WMMA operand layout)

    const v8f vzero = {0.f,0.f,0.f,0.f,0.f,0.f,0.f,0.f};

    // persistent across the grid-stride loop: num fragments + den partials
    v8f acc[2][4];
#pragma unroll
    for (int j = 0; j < 2; ++j)
#pragma unroll
        for (int d = 0; d < 4; ++d) acc[j][d] = vzero;
    float den_acc = 0.0f;

    // K == TPB == 512: one c2 element per thread
    c2s[t] = c2g[t];

    // GEMM1 wave mapping: 4 row-tiles x 4 column-groups (8 col-tiles each)
    const int rt = wave & 3;
    const int cg = wave >> 2;

    for (int tile = blockIdx.x; tile < NPTS / TN; tile += gridDim.x) {
        const int n0 = tile * TN;

        // ---------- build phase: inv_vn + transposed data_w tile ----------
        if (t < TN) ivn[t] = 1.0f / var_noise[n0 + t];
        {
            const int n     = t >> 3;            // 8 threads per row
            const int dbase = (t & 7) * 8;
            const float iv  = 1.0f / var_noise[n0 + n];
            const float* rp = data + (size_t)(n0 + n) * DIM + dbase;
            f32x4 q0 = *(const f32x4*)rp;
            f32x4 q1 = *(const f32x4*)(rp + 4);
#pragma unroll
            for (int i = 0; i < 4; ++i) {
                dataT[(dbase + i)     * DATAT_STRIDE + n] = pack_split(q0[i] * iv);
                dataT[(dbase + 4 + i) * DATAT_STRIDE + n] = pack_split(q1[i] * iv);
            }
        }
        __syncthreads();

        // ---------- GEMM1: logits = inv_vn * (x . c - 0.5*|c|^2) ----------
        // A-frags (data rows, f32 -> bf16 hi/lo). A-layout for 16x16x32:
        // elem i -> K = lh*16 + i (matches B)
        v16bf ahi[2], alo[2];
        {
            const int ng = n0 + rt * 16 + m;
            const float* rp = data + (size_t)ng * DIM;
#pragma unroll
            for (int kc = 0; kc < 2; ++kc) {
                const float* p = rp + kc * 32 + lh * 16;
                f32x4 q0 = *(const f32x4*)(p);
                f32x4 q1 = *(const f32x4*)(p + 4);
                f32x4 q2 = *(const f32x4*)(p + 8);
                f32x4 q3 = *(const f32x4*)(p + 12);
                v16us h, l;
#pragma unroll
                for (int i = 0; i < 4; ++i) {
                    unsigned short hb;
                    float x;
                    x = q0[i]; hb = bf16_trunc(x); h[i]      = hb; l[i]      = bf16_trunc(x - bf16_f32(hb));
                    x = q1[i]; hb = bf16_trunc(x); h[4 + i]  = hb; l[4 + i]  = bf16_trunc(x - bf16_f32(hb));
                    x = q2[i]; hb = bf16_trunc(x); h[8 + i]  = hb; l[8 + i]  = bf16_trunc(x - bf16_f32(hb));
                    x = q3[i]; hb = bf16_trunc(x); h[12 + i] = hb; l[12 + i] = bf16_trunc(x - bf16_f32(hb));
                }
                ahi[kc] = __builtin_bit_cast(v16bf, h);
                alo[kc] = __builtin_bit_cast(v16bf, l);
            }
        }
        float ivnv[8];
#pragma unroll
        for (int e = 0; e < 8; ++e) ivnv[e] = ivn[rt * 16 + e + 8 * lh];

        // keep rolled: limits live B-fragments, prevents scratch spills
#pragma unroll 1
        for (int cti = 0; cti < 8; ++cti) {
            const int ct = cg * 8 + cti;
            const int cl = ct * 16 + m;           // cluster column for this lane
            v8f a = vzero;
#pragma unroll
            for (int kc = 0; kc < 2; ++kc) {
                // B-layout: elem i -> K = lh*16 + i : contiguous 32B per lane
                const unsigned short* bp = cent_hi + cl * DIM + kc * 32 + lh * 16;
                const unsigned short* bq = cent_lo + cl * DIM + kc * 32 + lh * 16;
                v16bf bh = __builtin_bit_cast(v16bf, *(const v8u*)bp);
                v16bf bl = __builtin_bit_cast(v16bf, *(const v8u*)bq);
                a = __builtin_amdgcn_wmma_f32_16x16x32_bf16(false, ahi[kc], false, bh, (short)0, a, false, false);
                a = __builtin_amdgcn_wmma_f32_16x16x32_bf16(false, ahi[kc], false, bl, (short)0, a, false, false);
                a = __builtin_amdgcn_wmma_f32_16x16x32_bf16(false, alo[kc], false, bh, (short)0, a, false, false);
            }
            const float c2h = 0.5f * c2s[ct * 16 + m];
#pragma unroll
            for (int e = 0; e < 8; ++e) {
                const int row = rt * 16 + e + 8 * lh;   // C/D layout: M = e + 8*half
                logits[row * LOGITS_STRIDE + ct * 16 + m] = ivnv[e] * (a[e] - c2h);
            }
        }
        __syncthreads();

        // ---------- softmax over K=512 per row (vectorized LDS sweeps) ----
        {
            const int n = t >> 3, seg = t & 7;
            const f32x4* lp = (const f32x4*)(logits + n * LOGITS_STRIDE + seg * 64);
            float pm = -3.402823466e38f;
#pragma unroll 4
            for (int k = 0; k < 16; ++k) {
                f32x4 q = lp[k];
                pm = fmaxf(pm, fmaxf(fmaxf(q[0], q[1]), fmaxf(q[2], q[3])));
            }
            red[n * 8 + seg] = pm;
        }
        __syncthreads();
        if (t < TN) {
            float pm = red[t * 8];
#pragma unroll
            for (int s = 1; s < 8; ++s) pm = fmaxf(pm, red[t * 8 + s]);
            mrow[t] = pm;
        }
        __syncthreads();
        {
            const int n = t >> 3, seg = t & 7;
            f32x4* lp = (f32x4*)(logits + n * LOGITS_STRIDE + seg * 64);
            const float mb = mrow[n];
            float ps = 0.0f;
#pragma unroll 4
            for (int k = 0; k < 16; ++k) {
                f32x4 q = lp[k];
                f32x4 ev;
#pragma unroll
                for (int i = 0; i < 4; ++i) { ev[i] = __expf(q[i] - mb); ps += ev[i]; }
                lp[k] = ev;                       // overwrite logits with exp
            }
            red[n * 8 + seg] = ps;
        }
        __syncthreads();
        if (t < TN) {
            float s = 0.0f;
#pragma unroll
            for (int sg = 0; sg < 8; ++sg) s += red[t * 8 + sg];
            rrow[t] = 1.0f / s;
        }
        __syncthreads();

        // ---------- resp (k-major): transpose + den partials ----------
        {
            for (int n = 0; n < TN; ++n) {
                float r = logits[n * LOGITS_STRIDE + t] * rrow[n];
                den_acc += r * ivn[n];
                respT[t * RESPT_STRIDE + n] = pack_split(r);
            }
        }
        __syncthreads();

        // ---------- GEMM2: num += resp^T @ (data * inv_vn) ----------
        // nc stays rolled (halves live fragments); j/dt unrolled so the
        // persistent acc[][] register array is statically indexed.
#pragma unroll 1
        for (int nc = 0; nc < 2; ++nc) {
            v16bf rhi[2], rlo[2];
#pragma unroll
            for (int j = 0; j < 2; ++j) {
                const int krow = (wave * 2 + j) * 16 + m;
                const unsigned int* base = respT + krow * RESPT_STRIDE + nc * 32 + lh * 16;
                u32x4 r0 = *(const u32x4*)(base);
                u32x4 r1 = *(const u32x4*)(base + 4);
                u32x4 r2 = *(const u32x4*)(base + 8);
                u32x4 r3 = *(const u32x4*)(base + 12);
                v16us h, l;
#pragma unroll
                for (int i = 0; i < 4; ++i) {
                    unsigned int u;
                    u = r0[i]; h[i]      = (unsigned short)u; l[i]      = (unsigned short)(u >> 16);
                    u = r1[i]; h[4 + i]  = (unsigned short)u; l[4 + i]  = (unsigned short)(u >> 16);
                    u = r2[i]; h[8 + i]  = (unsigned short)u; l[8 + i]  = (unsigned short)(u >> 16);
                    u = r3[i]; h[12 + i] = (unsigned short)u; l[12 + i] = (unsigned short)(u >> 16);
                }
                rhi[j] = __builtin_bit_cast(v16bf, h);
                rlo[j] = __builtin_bit_cast(v16bf, l);
            }
#pragma unroll
            for (int dt = 0; dt < 4; ++dt) {
                const int dcol = dt * 16 + m;
                const unsigned int* bb = dataT + dcol * DATAT_STRIDE + nc * 32 + lh * 16;
                u32x4 b0 = *(const u32x4*)(bb);
                u32x4 b1 = *(const u32x4*)(bb + 4);
                u32x4 b2 = *(const u32x4*)(bb + 8);
                u32x4 b3 = *(const u32x4*)(bb + 12);
                v16us h, l;
#pragma unroll
                for (int i = 0; i < 4; ++i) {
                    unsigned int u;
                    u = b0[i]; h[i]      = (unsigned short)u; l[i]      = (unsigned short)(u >> 16);
                    u = b1[i]; h[4 + i]  = (unsigned short)u; l[4 + i]  = (unsigned short)(u >> 16);
                    u = b2[i]; h[8 + i]  = (unsigned short)u; l[8 + i]  = (unsigned short)(u >> 16);
                    u = b3[i]; h[12 + i] = (unsigned short)u; l[12 + i] = (unsigned short)(u >> 16);
                }
                v16bf xh = __builtin_bit_cast(v16bf, h);
                v16bf xl = __builtin_bit_cast(v16bf, l);
#pragma unroll
                for (int j = 0; j < 2; ++j) {
                    v8f a = acc[j][dt];
                    a = __builtin_amdgcn_wmma_f32_16x16x32_bf16(false, rhi[j], false, xh, (short)0, a, false, false);
                    a = __builtin_amdgcn_wmma_f32_16x16x32_bf16(false, rhi[j], false, xl, (short)0, a, false, false);
                    a = __builtin_amdgcn_wmma_f32_16x16x32_bf16(false, rlo[j], false, xh, (short)0, a, false, false);
                    acc[j][dt] = a;
                }
            }
        }
        __syncthreads();
    } // grid-stride tile loop

    // ---------- flush partial results (L2-resident atomics) ----------
#pragma unroll
    for (int j = 0; j < 2; ++j) {
        const int ktbase = (wave * 2 + j) * 16;
#pragma unroll
        for (int dt = 0; dt < 4; ++dt) {
#pragma unroll
            for (int e = 0; e < 8; ++e) {
                const int krow = ktbase + e + 8 * lh;
                const int dcol = dt * 16 + m;
                atomicAdd(num + krow * DIM + dcol, acc[j][dt][e]);
            }
        }
    }
    atomicAdd(den + t, den_acc);
}

// ---------- helpers ----------
__global__ void prep_cent(const float* __restrict__ cin,
                          unsigned short* __restrict__ chi,
                          unsigned short* __restrict__ clo,
                          float* __restrict__ c2)
{
    __shared__ float s[DIM];
    const int k = blockIdx.x;
    const int d = threadIdx.x;
    float c = cin[k * DIM + d];
    unsigned short h = bf16_trunc(c);
    chi[k * DIM + d] = h;
    clo[k * DIM + d] = bf16_trunc(c - bf16_f32(h));
    s[d] = c * c;
    __syncthreads();
    if (d == 0) {
        float tt = 0.0f;
        for (int i = 0; i < DIM; ++i) tt += s[i];
        c2[k] = tt;
    }
}

__global__ void zero_nd(float* __restrict__ num, float* __restrict__ den)
{
    int i = blockIdx.x * blockDim.x + threadIdx.x;
    if (i < KCL * DIM) num[i] = 0.0f;
    if (i < KCL) den[i] = 0.0f;
}

__global__ void finalize_cent(const float* __restrict__ num,
                              const float* __restrict__ den,
                              float* __restrict__ out)
{
    int i = blockIdx.x * blockDim.x + threadIdx.x;
    if (i < KCL * DIM) {
        out[i] = num[i] / (den[i >> 6] + 1.0e-8f);   // + 1/VAR_PRIOR
    }
}

// ---------- host launcher ----------
extern "C" void kernel_launch(void* const* d_in, const int* in_sizes, int n_in,
                              void* d_out, int out_size, void* d_ws, size_t ws_size,
                              hipStream_t stream)
{
    const float* data  = (const float*)d_in[0];
    const float* vn    = (const float*)d_in[1];
    const float* cinit = (const float*)d_in[2];
    float* out = (float*)d_out;

    char* ws = (char*)d_ws;
    unsigned short* chi = (unsigned short*)(ws);            // 65536 B
    unsigned short* clo = (unsigned short*)(ws + 65536);    // 65536 B
    float* c2  = (float*)(ws + 131072);                     // 2048 B
    float* num = (float*)(ws + 133120);                     // 131072 B
    float* den = (float*)(ws + 264192);                     // 2048 B

    hipFuncSetAttribute(reinterpret_cast<const void*>(em_main),
                        hipFuncAttributeMaxDynamicSharedMemorySize, LDS_BYTES);

    for (int s = 0; s < NSTEPS; ++s) {
        const float* src = (s == 0) ? cinit : (const float*)out;  // rolling centroids
        prep_cent<<<KCL, DIM, 0, stream>>>(src, chi, clo, c2);
        zero_nd<<<(KCL * DIM + 255) / 256, 256, 0, stream>>>(num, den);
        em_main<<<GRID_MAIN, TPB, LDS_BYTES, stream>>>(data, vn, chi, clo, c2, num, den);
        finalize_cent<<<(KCL * DIM + 255) / 256, 256, 0, stream>>>(num, den, out);
    }
    (void)in_sizes; (void)n_in; (void)out_size; (void)ws_size;
}
